// GatedRGCNClassifier_CNN_Improved_62216896250267
// MI455X (gfx1250) — matrix operand order repaired and checked by reading, not compiled, not run on hardware
//
#include <hip/hip_runtime.h>

// ---------------------------------------------------------------------------
// GatedRGCNClassifier for MI455X (gfx1250, wave32, WMMA)
// Dense GEMMs -> v_wmma_f32_16x16x32_bf16. Block stages a 16xK A-strip in LDS
// (conflict-free stride), 4 waves/block each compute a 16x64 tile, streaming
// K-major (pre-transposed) bf16 weights from global with b128 loads+prefetch.
// Graph scatter/gather, LN+GELU, segment softmax -> memory-bound helpers.
// ---------------------------------------------------------------------------

typedef __attribute__((ext_vector_type(16))) __bf16 bf16x16;
typedef __attribute__((ext_vector_type(8)))  float  f32x8;

#define KD_IN 784
#define KD_PAD 800     // 784 padded up to a multiple of 32 for WMMA K-loop
#define KD    768
#define KR    8
#define KL    3
#define KB    64
#define KN    32768
#define KE    262144

#define KMAX_STRIDE 808          // max K (800) + 8 pad: conflict-free LDS rows

// ------------------------------ WMMA GEMM ----------------------------------
// C[M,N] = A[M,K](bf16,row-major,lda) * B[K,N], with B supplied TRANSPOSED as
// Bt[N,K] (bf16, row-major, ldbt=K). fp32 out.
// blockIdx.y = M-tile (16 rows), blockIdx.x = 256-column block (4 waves x 64).
// The block cooperatively stages A[16, 0..K] into LDS once, then each wave
// runs the K-loop with A fragments from LDS and B fragments from global.
//
// Fragment layouts used by this kernel (16-bit A 16x32, B 32x16, f32 C/D):
//   A: lane -> row M=lane&15; fragment elems 0..7  = K[16*half .. 16*half+7],
//                             elems 8..15 = K[16*half+8 .. 16*half+15]
//   B: lane -> col N=lane&15; fragment elems 0..15 = K[16*half .. 16*half+15]
//   D: VGPR v -> row M = v + 8*half, col = lane&15
__global__ void wmma_gemm_bf16(const __bf16* __restrict__ A, int lda,
                               const __bf16* __restrict__ Bt, int ldbt,
                               float* __restrict__ C, int ldc,
                               int N, int K) {
  __shared__ __bf16 As[16 * KMAX_STRIDE];
  const int stride = K + 8;                  // dword-stride % 64 banks is coprime-ish
  const int mtile = blockIdx.y;
  const int tid   = threadIdx.x;

  // ---- cooperative stage of A strip: 16 rows x K elems, 16B chunks ----
  {
    const int chunksPerRow = K >> 3;         // K / 8 elems per 16B chunk
    const int totalChunks  = 16 * chunksPerRow;
    const __bf16* Abase = A + (size_t)(mtile * 16) * lda;
    for (int c = tid; c < totalChunks; c += 128) {
      const int row = c / chunksPerRow;
      const int off = (c % chunksPerRow) << 3;
      *(uint4*)(As + row * stride + off) = *(const uint4*)(Abase + (size_t)row * lda + off);
    }
  }
  __syncthreads();

  const int lane = tid & 31;
  const int wv   = tid >> 5;
  const int colbase = blockIdx.x * 256 + wv * 64;
  if (colbase >= N) return;                  // tail waves (after the barrier)

  const int half = lane >> 4;
  const int col0 = colbase + (lane & 15);

  const __bf16* asrow = As + (lane & 15) * stride;
  const __bf16* b0 = Bt + (size_t)(col0)      * ldbt;
  const __bf16* b1 = Bt + (size_t)(col0 + 16) * ldbt;
  const __bf16* b2 = Bt + (size_t)(col0 + 32) * ldbt;
  const __bf16* b3 = Bt + (size_t)(col0 + 48) * ldbt;

  f32x8 acc0 = {}, acc1 = {}, acc2 = {}, acc3 = {};

  for (int k0 = 0; k0 < K; k0 += 32) {
    if (k0 + 32 < K) {                       // stream-ahead (global_prefetch_b8)
      __builtin_prefetch(b0 + k0 + 32, 0, 1);
      __builtin_prefetch(b1 + k0 + 32, 0, 1);
      __builtin_prefetch(b2 + k0 + 32, 0, 1);
      __builtin_prefetch(b3 + k0 + 32, 0, 1);
    }
    union F { bf16x16 v; uint4 q[2]; };
    F a, f0, f1, f2, f3;
    a.q[0]  = *(const uint4*)(asrow + k0 + 16 * half);       // ds_load_b128
    a.q[1]  = *(const uint4*)(asrow + k0 + 16 * half + 8);
    f0.q[0] = *(const uint4*)(b0 + k0 + 16 * half);
    f0.q[1] = *(const uint4*)(b0 + k0 + 16 * half + 8);
    f1.q[0] = *(const uint4*)(b1 + k0 + 16 * half);
    f1.q[1] = *(const uint4*)(b1 + k0 + 16 * half + 8);
    f2.q[0] = *(const uint4*)(b2 + k0 + 16 * half);
    f2.q[1] = *(const uint4*)(b2 + k0 + 16 * half + 8);
    f3.q[0] = *(const uint4*)(b3 + k0 + 16 * half);
    f3.q[1] = *(const uint4*)(b3 + k0 + 16 * half + 8);
    acc0 = __builtin_amdgcn_wmma_f32_16x16x32_bf16(false, a.v, false, f0.v,
                                                   (short)0, acc0, false, false);
    acc1 = __builtin_amdgcn_wmma_f32_16x16x32_bf16(false, a.v, false, f1.v,
                                                   (short)0, acc1, false, false);
    acc2 = __builtin_amdgcn_wmma_f32_16x16x32_bf16(false, a.v, false, f2.v,
                                                   (short)0, acc2, false, false);
    acc3 = __builtin_amdgcn_wmma_f32_16x16x32_bf16(false, a.v, false, f3.v,
                                                   (short)0, acc3, false, false);
  }

#pragma unroll
  for (int v = 0; v < 8; ++v) {
    const size_t row = (size_t)(mtile * 16 + v + 8 * half) * ldc;
    C[row + col0]      = acc0[v];
    C[row + col0 + 16] = acc1[v];
    C[row + col0 + 32] = acc2[v];
    C[row + col0 + 48] = acc3[v];
  }
}

static inline void launch_gemm(const __bf16* A, int lda, const __bf16* Bt, int ldbt,
                               float* C, int ldc, int M, int N, int K,
                               hipStream_t s) {
  dim3 grid((N + 255) / 256, M / 16);
  wmma_gemm_bf16<<<grid, 128, 0, s>>>(A, lda, Bt, ldbt, C, ldc, N, K);
}

// --------------------------- conversion helpers ----------------------------
__device__ __forceinline__ float gelu_exact(float x) {
  return 0.5f * x * (1.f + erff(x * 0.70710678118654752f));
}

// fp32 -> bf16 copy (row-major preserved; used for A-side activations)
__global__ void f2bf_kernel(const float* __restrict__ in, __bf16* __restrict__ out, int n) {
  int i = blockIdx.x * blockDim.x + threadIdx.x;
  if (i < n) out[i] = (__bf16)in[i];
}

// batched transpose fp32[K,N] -> bf16[N,Kpad] (zero pad k >= K)
__global__ void transpose_pad_kernel(const float* __restrict__ in, __bf16* __restrict__ out,
                                     int K, int Kpad, int N, int nmat) {
  size_t i = (size_t)blockIdx.x * blockDim.x + threadIdx.x;
  size_t total = (size_t)nmat * N * Kpad;
  if (i >= total) return;
  int m = (int)(i / ((size_t)N * Kpad));
  size_t r = i % ((size_t)N * Kpad);
  int n = (int)(r / Kpad);
  int k = (int)(r % Kpad);
  float v = (k < K) ? in[(size_t)m * K * N + (size_t)k * N + n] : 0.f;
  out[i] = (__bf16)v;
}

// build padded bf16 input X = [type_vec | code_vec | 0-pad] : [N, 800]
__global__ void build_xb_kernel(const float* __restrict__ tv, const float* __restrict__ cv,
                                __bf16* __restrict__ xb) {
  int n = blockIdx.x;
  for (int k = threadIdx.x; k < KD_PAD; k += blockDim.x) {
    float v = (k < 16) ? tv[(size_t)n * 16 + k]
            : (k < KD_IN) ? cv[(size_t)n * KD + (k - 16)] : 0.f;
    xb[(size_t)n * KD_PAD + k] = (__bf16)v;
  }
}

// row-wise LayerNorm(+bias) + exact GELU; optional fp32 and bf16 outputs
__global__ void ln_gelu_kernel(const float* __restrict__ X, const float* __restrict__ bias,
                               const float* __restrict__ g, const float* __restrict__ b,
                               float* __restrict__ outF, __bf16* __restrict__ outB, int Dm) {
  __shared__ float s1[256], s2[256];
  const int row = blockIdx.x, tid = threadIdx.x;
  const float* x = X + (size_t)row * Dm;
  float s = 0.f, q = 0.f;
  for (int i = tid; i < Dm; i += 256) {
    float v = x[i] + (bias ? bias[i] : 0.f);
    s += v; q += v * v;
  }
  s1[tid] = s; s2[tid] = q; __syncthreads();
  for (int off = 128; off > 0; off >>= 1) {
    if (tid < off) { s1[tid] += s1[tid + off]; s2[tid] += s2[tid + off]; }
    __syncthreads();
  }
  const float mean = s1[0] / Dm;
  const float var  = s2[0] / Dm - mean * mean;
  const float rstd = rsqrtf(var + 1e-5f);
  for (int i = tid; i < Dm; i += 256) {
    float v = x[i] + (bias ? bias[i] : 0.f);
    float y = (v - mean) * rstd * g[i] + b[i];
    float ge = gelu_exact(y);
    if (outF) outF[(size_t)row * Dm + i] = ge;
    if (outB) outB[(size_t)row * Dm + i] = (__bf16)ge;
  }
}

// in-place bias + GELU (gate MLP hidden)
__global__ void gelu_bias_kernel(float* __restrict__ x, const float* __restrict__ bias,
                                 int total, int Dm) {
  int i = blockIdx.x * blockDim.x + threadIdx.x;
  if (i >= total) return;
  x[i] = gelu_exact(x[i] + bias[i % Dm]);
}

// ----------------------------- graph kernels -------------------------------
// agg[dst[e]] += tmp[src[e]] for edges of relation `rel`
__global__ void scatter_rel_kernel(const float* __restrict__ tmp,
                                   const int* __restrict__ src, const int* __restrict__ dst,
                                   const int* __restrict__ etype, int rel,
                                   float* __restrict__ agg) {
  const int e = blockIdx.x;
  if (etype[e] != rel) return;
  const float* ts = tmp + (size_t)src[e] * KD;
  float* ad = agg + (size_t)dst[e] * KD;
  for (int i = threadIdx.x; i < KD; i += blockDim.x) atomicAdd(&ad[i], ts[i]);
}

// gate[n] = g1[n,:] . Wg2 + bg2   (K = 384, one wave per node)
__global__ void gate_dot_kernel(const float* __restrict__ g1, const float* __restrict__ Wg2,
                                const float* __restrict__ bg2, float* __restrict__ gate) {
  const int n = blockIdx.x, lane = threadIdx.x;
  const float* row = g1 + (size_t)n * (KD / 2);
  float p = 0.f;
  for (int j = lane; j < KD / 2; j += 32) p += row[j] * Wg2[j];
  for (int off = 16; off > 0; off >>= 1) p += __shfl_down(p, off);
  if (lane == 0) gate[n] = p + bg2[0];
}

__device__ __forceinline__ void atomicMaxF(float* addr, float val) {
  unsigned* ua = (unsigned*)addr;
  unsigned old = *ua;
  while (true) {
    if (__uint_as_float(old) >= val) break;
    unsigned prev = atomicCAS(ua, old, __float_as_uint(val));
    if (prev == old) break;
    old = prev;
  }
}

__global__ void pool_init_kernel(float* gmax, float* denom, float* grepr) {
  int i = blockIdx.x * blockDim.x + threadIdx.x;
  if (i < KB) { gmax[i] = -1e30f; denom[i] = 0.f; }
  if (i < KB * KD) grepr[i] = 0.f;
}

__global__ void pool_max_kernel(const float* __restrict__ gate, const int* __restrict__ ng,
                                float* __restrict__ gmax) {
  int n = blockIdx.x * blockDim.x + threadIdx.x;
  if (n < KN) atomicMaxF(&gmax[ng[n]], gate[n]);
}

__global__ void pool_exp_kernel(const float* __restrict__ gate, const int* __restrict__ ng,
                                const float* __restrict__ gmax, float* __restrict__ eexp,
                                float* __restrict__ denom) {
  int n = blockIdx.x * blockDim.x + threadIdx.x;
  if (n >= KN) return;
  float e = expf(gate[n] - gmax[ng[n]]);
  eexp[n] = e;
  atomicAdd(&denom[ng[n]], e);
}

__global__ void pool_acc_kernel(const float* __restrict__ h, const float* __restrict__ eexp,
                                const float* __restrict__ denom, const int* __restrict__ ng,
                                float* __restrict__ grepr) {
  const int n = blockIdx.x, g = ng[n];
  const float w = eexp[n] / denom[g];
  const float* hn = h + (size_t)n * KD;
  float* gr = grepr + (size_t)g * KD;
  for (int i = threadIdx.x; i < KD; i += blockDim.x) atomicAdd(&gr[i], w * hn[i]);
}

// classifier head: logits = z @ Wc2 + bc2, then log_softmax over 2 classes
__global__ void head_kernel(const float* __restrict__ z, const float* __restrict__ Wc2,
                            const float* __restrict__ bc2, float* __restrict__ out) {
  const int b = blockIdx.x, lane = threadIdx.x;
  const float* zr = z + (size_t)b * 512;
  float p0 = 0.f, p1 = 0.f;
  for (int j = lane; j < 512; j += 32) {
    float v = zr[j];
    p0 += v * Wc2[j * 2 + 0];
    p1 += v * Wc2[j * 2 + 1];
  }
  for (int off = 16; off > 0; off >>= 1) {
    p0 += __shfl_down(p0, off);
    p1 += __shfl_down(p1, off);
  }
  if (lane == 0) {
    float l0 = p0 + bc2[0], l1 = p1 + bc2[1];
    float m = fmaxf(l0, l1);
    float lse = m + logf(expf(l0 - m) + expf(l1 - m));
    out[b * 2 + 0] = l0 - lse;
    out[b * 2 + 1] = l1 - lse;
  }
}

// ------------------------------- launcher ----------------------------------
static inline char* carve(char*& cur, size_t bytes) {
  char* p = cur;
  cur += (bytes + 255) & ~(size_t)255;
  return p;
}

extern "C" void kernel_launch(void* const* d_in, const int* in_sizes, int n_in,
                              void* d_out, int out_size, void* d_ws, size_t ws_size,
                              hipStream_t stream) {
  (void)in_sizes; (void)n_in; (void)out_size; (void)ws_size;
  // ---- inputs (setup_inputs dict order) ----
  const float* type_vec = (const float*)d_in[0];
  const float* code_vec = (const float*)d_in[1];
  const float* W_in     = (const float*)d_in[2];
  const float* b_in     = (const float*)d_in[3];
  const float* ln_in_g  = (const float*)d_in[4];
  const float* ln_in_b  = (const float*)d_in[5];
  const float* W_rel    = (const float*)d_in[6];
  const float* W_loop   = (const float*)d_in[7];
  const float* rgcn_b   = (const float*)d_in[8];
  const float* ln_g     = (const float*)d_in[9];
  const float* ln_b     = (const float*)d_in[10];
  const float* Wg1      = (const float*)d_in[11];
  const float* bg1      = (const float*)d_in[12];
  const float* Wg2      = (const float*)d_in[13];
  const float* bg2      = (const float*)d_in[14];
  const float* Wc1      = (const float*)d_in[15];
  const float* bc1      = (const float*)d_in[16];
  const float* ln_c_g   = (const float*)d_in[17];
  const float* ln_c_b   = (const float*)d_in[18];
  const float* Wc2      = (const float*)d_in[19];
  const float* bc2      = (const float*)d_in[20];
  const int*   src      = (const int*)d_in[21];
  const int*   dst      = (const int*)d_in[22];
  const int*   etype    = (const int*)d_in[23];
  const int*   ngraph   = (const int*)d_in[24];
  float* out = (float*)d_out;

  // ---- workspace carving ----
  char* cur = (char*)d_ws;
  float*  h    = (float*) carve(cur, (size_t)KN * KD * 4);   // activations (fp32)
  __bf16* hB   = (__bf16*)carve(cur, (size_t)KN * KD * 2);   // activations (bf16)
  float*  agg  = (float*) carve(cur, (size_t)KN * KD * 4);   // message accumulator
  float*  tmp  = (float*) carve(cur, (size_t)KN * KD * 4);   // GEMM scratch / gate hidden
  // bf16 weights, TRANSPOSED to [N, K] (converted once per launch)
  const size_t nWin  = (size_t)KD * KD_PAD;          // [768, 800]
  const size_t nWrel = (size_t)KL * KR * KD * KD;    // 24 x [768, 768]
  const size_t nWlp  = (size_t)KL * KD * KD;         //  3 x [768, 768]
  const size_t nWg1  = (size_t)(KD / 2) * KD;        // [384, 768]
  const size_t nWc1  = (size_t)512 * KD;             // [512, 768]
  __bf16* wbf   = (__bf16*)carve(cur, (nWin + nWrel + nWlp + nWg1 + nWc1) * 2);
  __bf16* WinT  = wbf;
  __bf16* WrelT = WinT + nWin;
  __bf16* WloopT= WrelT + nWrel;
  __bf16* Wg1T  = WloopT + nWlp;
  __bf16* Wc1T  = Wg1T + nWg1;
  float*  gatev = (float*) carve(cur, (size_t)KN * 4);
  float*  eexp  = (float*) carve(cur, (size_t)KN * 4);
  float*  gmax  = (float*) carve(cur, (size_t)KB * 4);
  float*  denom = (float*) carve(cur, (size_t)KB * 4);
  float*  grepr = (float*) carve(cur, (size_t)KB * KD * 4);
  __bf16* greprB= (__bf16*)carve(cur, (size_t)KB * KD * 2);
  float*  zpre  = (float*) carve(cur, (size_t)KB * 512 * 4);
  float*  zf    = (float*) carve(cur, (size_t)KB * 512 * 4);
  __bf16* Xb    = (__bf16*)agg;  // alias: Xb only live before agg is first used

  // ---- weight conversion: fp32 [K,N] -> bf16 transposed [N,Kpad] ----
  auto tpose = [&](const float* s, __bf16* d, int K, int Kpad, int N, int nmat) {
    size_t total = (size_t)nmat * N * Kpad;
    transpose_pad_kernel<<<(int)((total + 255) / 256), 256, 0, stream>>>(s, d, K, Kpad, N, nmat);
  };
  tpose(W_in,   WinT,   KD_IN, KD_PAD, KD,     1);
  tpose(W_rel,  WrelT,  KD,    KD,     KD,     KL * KR);
  tpose(W_loop, WloopT, KD,    KD,     KD,     KL);
  tpose(Wg1,    Wg1T,   KD,    KD,     KD / 2, 1);
  tpose(Wc1,    Wc1T,   KD,    KD,     512,    1);

  // ---- input projection: h = gelu(LN(X @ W_in + b_in)) ----
  build_xb_kernel<<<KN, 256, 0, stream>>>(type_vec, code_vec, Xb);
  launch_gemm(Xb, KD_PAD, WinT, KD_PAD, tmp, KD, KN, KD, KD_PAD, stream);
  ln_gelu_kernel<<<KN, 256, 0, stream>>>(tmp, b_in, ln_in_g, ln_in_b, h, hB, KD);

  // ---- RGCN layers ----
  for (int l = 0; l < KL; ++l) {
    // agg = h @ W_loop[l]   (self-loop initializes the accumulator)
    launch_gemm(hB, KD, WloopT + (size_t)l * KD * KD, KD, agg, KD, KN, KD, KD, stream);
    for (int r = 0; r < KR; ++r) {
      // tmp = h @ W_rel[l,r]; then scatter-add over matching edges
      launch_gemm(hB, KD, WrelT + ((size_t)l * KR + r) * KD * KD, KD,
                  tmp, KD, KN, KD, KD, stream);
      scatter_rel_kernel<<<KE, 256, 0, stream>>>(tmp, src, dst, etype, r, agg);
    }
    ln_gelu_kernel<<<KN, 256, 0, stream>>>(agg, rgcn_b + (size_t)l * KD,
                                           ln_g + (size_t)l * KD, ln_b + (size_t)l * KD,
                                           h, hB, KD);
  }

  // ---- global attention pooling ----
  launch_gemm(hB, KD, Wg1T, KD, tmp, KD / 2, KN, KD / 2, KD, stream);
  gelu_bias_kernel<<<(KN * (KD / 2) + 255) / 256, 256, 0, stream>>>(tmp, bg1,
                                                                    KN * (KD / 2), KD / 2);
  gate_dot_kernel<<<KN, 32, 0, stream>>>(tmp, Wg2, bg2, gatev);
  pool_init_kernel<<<(KB * KD + 255) / 256, 256, 0, stream>>>(gmax, denom, grepr);
  pool_max_kernel<<<(KN + 255) / 256, 256, 0, stream>>>(gatev, ngraph, gmax);
  pool_exp_kernel<<<(KN + 255) / 256, 256, 0, stream>>>(gatev, ngraph, gmax, eexp, denom);
  pool_acc_kernel<<<KN, 256, 0, stream>>>(h, eexp, denom, ngraph, grepr);

  // ---- classifier head ----
  f2bf_kernel<<<(KB * KD + 255) / 256, 256, 0, stream>>>(grepr, greprB, KB * KD);
  launch_gemm(greprB, KD, Wc1T, KD, zpre, 512, KB, 512, KD, stream);
  ln_gelu_kernel<<<KB, 256, 0, stream>>>(zpre, bc1, ln_c_g, ln_c_b, zf, (__bf16*)nullptr, 512);
  head_kernel<<<KB, 32, 0, stream>>>(zf, Wc2, bc2, out);
}
